// DQN_9904194584789
// MI455X (gfx1250) — matrix operand order repaired and checked by reading, not compiled, not run on hardware
//
#include <hip/hip_runtime.h>

typedef __attribute__((ext_vector_type(16))) _Float16     v16h;
typedef __attribute__((ext_vector_type(8)))  float        v8f;
typedef __attribute__((ext_vector_type(4)))  unsigned int u32x4;
typedef __attribute__((ext_vector_type(8)))  int          i32x8;
typedef __attribute__((ext_vector_type(4)))  int          i32x4;
typedef __attribute__((ext_vector_type(4)))  float        f32x4;

// Problem constants
#define BB 16
#define AA 256
#define UU 64

union AFrag  { v16h h; u32x4 q[2]; };
union Pack8  { u32x4 q; _Float16 h[8]; };

// ---------------------------------------------------------------------------
// prep: adjacency (f16 0/1, row-major [b*A+a][a2]) + P=sum relu(w), N=sum relu(-w)
// one wave per row; each lane owns 8 consecutive columns -> b128 stores
// ---------------------------------------------------------------------------
__global__ __launch_bounds__(256) void prep_kernel(const float* __restrict__ x,
                                                   _Float16* __restrict__ adj,
                                                   float* __restrict__ P,
                                                   float* __restrict__ N) {
  int lane = threadIdx.x & 31;
  int row  = blockIdx.x * 8 + (threadIdx.x >> 5);   // 0..4095 = b*256 + a
  int b = row >> 8, a = row & 255;
  const f32x4* wrow = (const f32x4*)(x + ((size_t)(b * (AA + 2) + 2 + a)) * AA);
  f32x4 f0 = wrow[lane * 2 + 0];
  f32x4 f1 = wrow[lane * 2 + 1];
  float p = 0.f, n = 0.f;
  Pack8 st;
#pragma unroll
  for (int i = 0; i < 4; ++i) {
    float w0 = f0[i], w1 = f1[i];
    st.h[i]     = (w0 != 0.f) ? (_Float16)1.0f : (_Float16)0.0f;
    st.h[i + 4] = (w1 != 0.f) ? (_Float16)1.0f : (_Float16)0.0f;
    p += fmaxf(w0, 0.f) + fmaxf(w1, 0.f);
    n += fmaxf(-w0, 0.f) + fmaxf(-w1, 0.f);
  }
  *(u32x4*)(adj + (size_t)row * AA + lane * 8) = st.q;
  for (int off = 16; off > 0; off >>= 1) {
    p += __shfl_down(p, off, 32);
    n += __shfl_down(n, off, 32);
  }
  if (lane == 0) { P[row] = p; N[row] = n; }
}

// ---------------------------------------------------------------------------
// const: Cp/Cn vectors (1/A folded in) + W1 transposed to f16 ([v][u])
// ---------------------------------------------------------------------------
__global__ __launch_bounds__(64) void const_kernel(const float* __restrict__ W2,
                                                   const float* __restrict__ W3,
                                                   const float* __restrict__ W1,
                                                   float* __restrict__ Cp,
                                                   float* __restrict__ Cn,
                                                   _Float16* __restrict__ W1hT) {
  int v = threadIdx.x;
  float cp = 0.f, cn = 0.f;
  for (int u = 0; u < UU; ++u) {
    float w3 = W3[u], w2 = W2[u * UU + v];
    cp += fmaxf(w3, 0.f) * w2;
    cn += fmaxf(-w3, 0.f) * w2;
    W1hT[v * UU + u] = (_Float16)W1[u * UU + v];
  }
  Cp[v] = cp * (1.0f / AA);
  Cn[v] = cn * (1.0f / AA);
}

// ---------------------------------------------------------------------------
// e13 (transposed, [b][v][a]): e13T = cur_sol*W0[v] + P*Cp[v] + N*Cn[v]
// ---------------------------------------------------------------------------
__global__ __launch_bounds__(256) void e13_kernel(const float* __restrict__ x,
                                                  const float* __restrict__ W0,
                                                  const float* __restrict__ P,
                                                  const float* __restrict__ N,
                                                  const float* __restrict__ Cp,
                                                  const float* __restrict__ Cn,
                                                  float* __restrict__ e13T) {
  int idx = blockIdx.x * 256 + threadIdx.x;        // 0..262143  (b, v, a)
  int b = idx >> 14, v = (idx >> 8) & 63, a = idx & 255;
  float cur = x[((size_t)b * (AA + 2)) * AA + a];  // x[b,0,a]
  e13T[idx] = cur * W0[v] + P[b * AA + a] * Cp[v] + N[b * AA + a] * Cn[v];
}

// ---------------------------------------------------------------------------
// iteration 1: embedding = relu(e13); write transposed f16 ping buffer
// ---------------------------------------------------------------------------
__global__ __launch_bounds__(256) void emb_init_kernel(const float* __restrict__ e13T,
                                                       _Float16* __restrict__ embT16) {
  int idx = blockIdx.x * 256 + threadIdx.x;
  embT16[idx] = (_Float16)fmaxf(e13T[idx], 0.f);
}

// ---------------------------------------------------------------------------
// msg_pass: 4 waves per WG; wave w owns a 16-row tile.
//   Z   = adj(16x256) @ embT(256x64)ᵀ / A       (WMMA f16->f32, K=256)
//   e2  = Z(16x64) @ W1(64x64)                  (WMMA f16->f32, K=64)
//   out = relu(e13 + e2)  -> transposed f16 ping-pong (+ f32 on final pass)
// Per-batch embedding staged to LDS via the Tensor Data Mover when available.
// ---------------------------------------------------------------------------
__global__ __launch_bounds__(128) void msg_pass_kernel(const _Float16* __restrict__ adj,
                                                       const _Float16* __restrict__ embInT,
                                                       const _Float16* __restrict__ W1hT,
                                                       const float* __restrict__ e13T,
                                                       float* __restrict__ emb32,
                                                       _Float16* __restrict__ embOutT,
                                                       int writeF32) {
  __shared__ __align__(16) _Float16 sEmbT[UU * AA];     // 32 KB, [u][k]
  __shared__ __align__(16) _Float16 sW1T[UU * UU];      // 8 KB,  [v][u]
  __shared__ __align__(16) _Float16 sD1[4][16 * UU];    // 4 x 2 KB, [m][u]

  int tid  = threadIdx.x;
  int wv   = tid >> 5;
  int lane = tid & 31;
  int b    = blockIdx.y;
  int row0 = blockIdx.x * 64 + wv * 16;
  int m    = lane & 15;
  int half = lane >> 4;
  int n    = lane & 15;

  const _Float16* embB = embInT + (size_t)b * UU * AA;

#if __has_builtin(__builtin_amdgcn_tensor_load_to_lds) && __has_builtin(__builtin_amdgcn_s_wait_tensorcnt)
  if (wv == 0) {
    // 1-D TDM copy: 32 KB (4096 x 8B elements), global -> LDS
    unsigned lds  = (unsigned)(size_t)(const void*)&sEmbT[0];
    unsigned long long ga = (unsigned long long)(size_t)embB;
    u32x4 g0;
    g0[0] = 1u;                                             // count=1, user mode
    g0[1] = lds;                                            // lds_addr
    g0[2] = (unsigned)(ga & 0xFFFFFFFFu);                   // global_addr[31:0]
    g0[3] = (unsigned)((ga >> 32) & 0x01FFFFFFu) | (2u << 30); // addr[56:32], type=2
    i32x8 g1;
    g1[0] = (int)(3u << 16);          // data_size=3 (8B), no multicast/pad/iterate
    g1[1] = (int)((4096u & 0xFFFFu) << 16);  // tensor_dim0[15:0]
    g1[2] = (int)(1u << 16);          // tensor_dim0[31:16]=0, tensor_dim1 lo=1
    g1[3] = (int)(4096u << 16);       // tensor_dim1 hi=0, tile_dim0=4096
    g1[4] = 0;                        // tile_dim1=0 (unused), tile_dim2=0
    g1[5] = 4096;                     // tensor_dim0_stride[31:0]
    g1[6] = 0;
    g1[7] = 0;
    i32x4 g2 = {0, 0, 0, 0};
    i32x4 g3 = {0, 0, 0, 0};
    i32x8 g4 = {0, 0, 0, 0, 0, 0, 0, 0};
    __builtin_amdgcn_tensor_load_to_lds(g0, g1, g2, g3, g4, 0);
    __builtin_amdgcn_s_wait_tensorcnt((short)0);
  }
#else
  {
    const u32x4* src = (const u32x4*)embB;
    u32x4* dst = (u32x4*)sEmbT;
    for (int i = tid; i < (UU * AA * 2) / 16; i += 128) dst[i] = src[i];
  }
#endif
  {
    const u32x4* w1s = (const u32x4*)W1hT;
    u32x4* w1d = (u32x4*)sW1T;
    for (int i = tid; i < (UU * UU * 2) / 16; i += 128) w1d[i] = w1s[i];
  }
  __syncthreads();

  const _Float16* adjRow = adj + ((size_t)b * AA + row0 + m) * AA;
  __builtin_prefetch(adjRow, 0, 0);

  // Hoist all A-fragments (adjacency, reused across the 4 N-tiles)
  AFrag aF[8];
#pragma unroll
  for (int ks = 0; ks < 8; ++ks) {
    int k0 = ks * 32;
    aF[ks].q[0] = *(const u32x4*)(adjRow + k0 + half * 8);
    aF[ks].q[1] = *(const u32x4*)(adjRow + k0 + 16 + half * 8);
  }

  // ---- GEMM1: Z = adj @ emb (emb transposed in LDS -> contiguous B-frags) ----
  _Float16* sD1w = &sD1[wv][0];
#pragma unroll
  for (int nt = 0; nt < 4; ++nt) {
    int n0 = nt * 16;
    const _Float16* bcol = sEmbT + (size_t)(n0 + n) * AA + half * 8;
    v8f acc = {};
#pragma unroll
    for (int ks = 0; ks < 8; ++ks) {
      AFrag bF;
      bF.q[0] = *(const u32x4*)(bcol + ks * 32);
      bF.q[1] = *(const u32x4*)(bcol + ks * 32 + 16);
      acc = __builtin_amdgcn_wmma_f32_16x16x32_f16(false, aF[ks].h, false, bF.h,
                                                   (short)0, acc, false, false);
    }
#pragma unroll
    for (int i = 0; i < 8; ++i)
      sD1w[(half * 8 + i) * UU + n0 + n] = (_Float16)(acc[i] * (1.0f / AA));
  }
  __syncthreads();

  // ---- GEMM2: e2 = Z @ W1 (W1 transposed in LDS), fused relu(e13+e2) ----
#pragma unroll
  for (int nt = 0; nt < 4; ++nt) {
    int n0 = nt * 16;
    int col = n0 + n;
    v8f acc = {};
#pragma unroll
    for (int ks = 0; ks < 2; ++ks) {
      int k0 = ks * 32;
      AFrag af, bf;
      af.q[0] = *(const u32x4*)(sD1w + m * UU + k0 + half * 8);
      af.q[1] = *(const u32x4*)(sD1w + m * UU + k0 + 16 + half * 8);
      const _Float16* wcol = sW1T + (size_t)col * UU + k0 + half * 8;
      bf.q[0] = *(const u32x4*)(wcol);
      bf.q[1] = *(const u32x4*)(wcol + 16);
      acc = __builtin_amdgcn_wmma_f32_16x16x32_f16(false, af.h, false, bf.h,
                                                   (short)0, acc, false, false);
    }
    size_t tbase = ((size_t)b * UU + col) * AA + row0 + half * 8;
    const f32x4* ep = (const f32x4*)(e13T + tbase);
    f32x4 e0 = ep[0], e1 = ep[1];
    float r[8];
#pragma unroll
    for (int i = 0; i < 4; ++i) {
      r[i]     = fmaxf(acc[i] + e0[i], 0.f);
      r[i + 4] = fmaxf(acc[i + 4] + e1[i], 0.f);
    }
    Pack8 st;
#pragma unroll
    for (int i = 0; i < 8; ++i) st.h[i] = (_Float16)r[i];
    *(u32x4*)(embOutT + tbase) = st.q;
    if (writeF32) {
#pragma unroll
      for (int i = 0; i < 8; ++i)
        emb32[(size_t)(b * AA + row0 + half * 8 + i) * UU + col] = r[i];
    }
  }
}

// ---------------------------------------------------------------------------
// heads: per-row dueling MLP; one 64-thread block per (b,a) row
// ---------------------------------------------------------------------------
__global__ __launch_bounds__(64) void heads_kernel(const float* __restrict__ emb32,
                                                   const float* __restrict__ aw1,
                                                   const float* __restrict__ ab1,
                                                   const float* __restrict__ aw2,
                                                   const float* __restrict__ ab2,
                                                   const float* __restrict__ aw3,
                                                   const float* __restrict__ ab3,
                                                   const float* __restrict__ vw1,
                                                   const float* __restrict__ vb1,
                                                   const float* __restrict__ vw2,
                                                   const float* __restrict__ vb2,
                                                   float* __restrict__ outA,
                                                   float* __restrict__ outV) {
  __shared__ float sE[64], sH1[64], sHv[64], sH2[32];
  int row = blockIdx.x, j = threadIdx.x;
  sE[j] = emb32[(size_t)row * UU + j];
  __syncthreads();
  float aAcc = ab1[j], vAcc = vb1[j];
#pragma unroll 8
  for (int u = 0; u < 64; ++u) {
    float e = sE[u];
    aAcc += e * aw1[u * 64 + j];
    vAcc += e * vw1[u * 64 + j];
  }
  sH1[j] = fmaxf(aAcc, 0.f);
  sHv[j] = fmaxf(vAcc, 0.f);
  __syncthreads();
  if (j < 32) {
    float acc = ab2[j];
#pragma unroll 8
    for (int u = 0; u < 64; ++u) acc += sH1[u] * aw2[u * 32 + j];
    sH2[j] = fmaxf(acc, 0.f);
  }
  __syncthreads();
  if (j == 0) {
    float oa = ab3[0];
#pragma unroll 8
    for (int u = 0; u < 32; ++u) oa += sH2[u] * aw3[u];
    float ov = vb2[0];
#pragma unroll 8
    for (int u = 0; u < 64; ++u) ov += sHv[u] * vw2[u];
    outA[row] = oa;
    outV[row] = ov;
  }
}

// ---------------------------------------------------------------------------
// finalize: deterministic per-batch tree reduction + dueling combine + mask*10
// ---------------------------------------------------------------------------
__global__ __launch_bounds__(256) void finalize_kernel(const float* __restrict__ x,
                                                       const float* __restrict__ outA,
                                                       const float* __restrict__ outV,
                                                       float* __restrict__ out) {
  __shared__ float sA[256], sV[256];
  int b = blockIdx.x, t = threadIdx.x;
  float a = outA[b * AA + t], v = outV[b * AA + t];
  sA[t] = a; sV[t] = v;
  __syncthreads();
  for (int off = 128; off > 0; off >>= 1) {
    if (t < off) { sA[t] += sA[t + off]; sV[t] += sV[t + off]; }
    __syncthreads();
  }
  float sumA = sA[0], sumV = sV[0];
  float mask = x[((size_t)b * (AA + 2) + 1) * AA + t];   // x[b,1,t]
  out[b * AA + t] = sumV + a - sumA + mask * 10.0f;
}

// ---------------------------------------------------------------------------
extern "C" void kernel_launch(void* const* d_in, const int* in_sizes, int n_in,
                              void* d_out, int out_size, void* d_ws, size_t ws_size,
                              hipStream_t stream) {
  (void)in_sizes; (void)n_in; (void)out_size; (void)ws_size;
  const float* x   = (const float*)d_in[0];
  const float* W0  = (const float*)d_in[1];
  const float* W1  = (const float*)d_in[2];
  const float* W2  = (const float*)d_in[3];
  const float* W3  = (const float*)d_in[4];
  const float* aw1 = (const float*)d_in[5];
  const float* ab1 = (const float*)d_in[6];
  const float* aw2 = (const float*)d_in[7];
  const float* ab2 = (const float*)d_in[8];
  const float* aw3 = (const float*)d_in[9];
  const float* ab3 = (const float*)d_in[10];
  const float* vw1 = (const float*)d_in[11];
  const float* vb1 = (const float*)d_in[12];
  const float* vw2 = (const float*)d_in[13];
  const float* vb2 = (const float*)d_in[14];
  float* out = (float*)d_out;

  char* ws = (char*)d_ws;
  size_t off = 0;
  auto alloc = [&](size_t bytes) -> void* {
    void* p = ws + off;
    off = (off + bytes + 255) & ~(size_t)255;
    return p;
  };
  _Float16* adj   = (_Float16*)alloc((size_t)BB * AA * AA * 2);   // 2 MB
  float*    P     = (float*)alloc((size_t)BB * AA * 4);
  float*    N     = (float*)alloc((size_t)BB * AA * 4);
  float*    Cp    = (float*)alloc(UU * 4);
  float*    Cn    = (float*)alloc(UU * 4);
  _Float16* W1hT  = (_Float16*)alloc(UU * UU * 2);
  float*    e13T  = (float*)alloc((size_t)BB * AA * UU * 4);      // 1 MB
  float*    emb32 = (float*)alloc((size_t)BB * AA * UU * 4);      // 1 MB
  _Float16* embA  = (_Float16*)alloc((size_t)BB * AA * UU * 2);   // 512 KB (transposed)
  _Float16* embB  = (_Float16*)alloc((size_t)BB * AA * UU * 2);   // 512 KB (transposed)
  float*    outA  = (float*)alloc((size_t)BB * AA * 4);
  float*    outV  = (float*)alloc((size_t)BB * AA * 4);

  prep_kernel<<<(BB * AA) / 8, 256, 0, stream>>>(x, adj, P, N);
  const_kernel<<<1, 64, 0, stream>>>(W2, W3, W1, Cp, Cn, W1hT);
  e13_kernel<<<(BB * AA * UU) / 256, 256, 0, stream>>>(x, W0, P, N, Cp, Cn, e13T);

  // T = 3 iterations: iteration 1 is relu(e13); then two WMMA message passes.
  emb_init_kernel<<<(BB * AA * UU) / 256, 256, 0, stream>>>(e13T, embA);
  dim3 g(AA / 64, BB);
  msg_pass_kernel<<<g, 128, 0, stream>>>(adj, embA, W1hT, e13T, emb32, embB, 0);
  msg_pass_kernel<<<g, 128, 0, stream>>>(adj, embB, W1hT, e13T, emb32, embA, 1);

  heads_kernel<<<BB * AA, 64, 0, stream>>>(emb32, aw1, ab1, aw2, ab2, aw3, ab3,
                                           vw1, vb1, vw2, vb2, outA, outV);
  finalize_kernel<<<BB, 256, 0, stream>>>(x, outA, outV, out);
}
